// LSTQuant_13649406067318
// MI455X (gfx1250) — matrix operand order repaired and checked by reading, not compiled, not run on hardware
//
#include <hip/hip_runtime.h>
#include <stdint.h>

#define LAYERS 8
#define NEXP   8
#define RANK   64
#define HDIM   2048
#define TOK    (4 * 2048)     // 8192 tokens
#define SCALING 0.5f

typedef __attribute__((ext_vector_type(16))) __bf16 v16bf;
typedef __attribute__((ext_vector_type(8)))  float  v8f;

union Frag { uint4 q[2]; v16bf v; };

__device__ __forceinline__ uint16_t f2bf(float f) {
    uint32_t u = __float_as_uint(f);
    u += 0x7FFFu + ((u >> 16) & 1u);      // round-to-nearest-even
    return (uint16_t)(u >> 16);
}

__device__ __forceinline__ v8f wmma_bf16(const Frag& a, const Frag& b, v8f c) {
    return __builtin_amdgcn_wmma_f32_16x16x32_bf16(false, a.v, false, b.v,
                                                   (short)0, c, false, false);
}

// ---------------------------------------------------------------- weights fp32 -> bf16
__global__ void cvt_bf16(const float* __restrict__ src, uint16_t* __restrict__ dst, int n) {
    int i = blockIdx.x * blockDim.x + threadIdx.x;
    int stride = gridDim.x * blockDim.x;
    for (; i < n; i += stride) dst[i] = f2bf(src[i]);
}

// ---------------------------------------------------------------- blend + gate + route
// one wave32 per token; float4-vectorized streaming
__global__ __launch_bounds__(256)
void blend_gate_route(const float* __restrict__ xsrc, const float* __restrict__ hx,
                      const float* __restrict__ gates, int layer,
                      const float* __restrict__ gw,        // [E,H] this layer
                      uint16_t* __restrict__ Xb, float* __restrict__ coeff,
                      int* __restrict__ order, int* __restrict__ counts)
{
    const int lane = threadIdx.x & 31;
    const int wave = threadIdx.x >> 5;
    const int t = blockIdx.x * (blockDim.x >> 5) + wave;

    const float g = gates[layer];
    const float z = 1.0f / (1.0f + __expf(-g));
    const float o = 1.0f - z;

    float acc[NEXP];
#pragma unroll
    for (int e = 0; e < NEXP; ++e) acc[e] = 0.0f;

    const float* xr = xsrc + (size_t)t * HDIM;
    const float* hr = hx   + (size_t)t * HDIM;
    uint16_t*    xb = Xb   + (size_t)t * HDIM;

    for (int k = lane * 4; k < HDIM; k += 128) {
        __builtin_prefetch(xr + k + 1024, 0, 0);   // global_prefetch_b8
        __builtin_prefetch(hr + k + 1024, 0, 0);
        float4 xv = *(const float4*)(xr + k);
        float4 hv = *(const float4*)(hr + k);
        float v0 = z * xv.x + o * hv.x;
        float v1 = z * xv.y + o * hv.y;
        float v2 = z * xv.z + o * hv.z;
        float v3 = z * xv.w + o * hv.w;
        uint2 pk;
        pk.x = (uint32_t)f2bf(v0) | ((uint32_t)f2bf(v1) << 16);
        pk.y = (uint32_t)f2bf(v2) | ((uint32_t)f2bf(v3) << 16);
        *(uint2*)(xb + k) = pk;
#pragma unroll
        for (int e = 0; e < NEXP; ++e) {
            float4 g4 = *(const float4*)(gw + e * HDIM + k);
            acc[e] += v0 * g4.x + v1 * g4.y + v2 * g4.z + v3 * g4.w;
        }
    }

    // wave32 cross-lane reduction
#pragma unroll
    for (int e = 0; e < NEXP; ++e) {
#pragma unroll
        for (int off = 16; off; off >>= 1) acc[e] += __shfl_xor(acc[e], off, 32);
    }

    // softmax / top-1 (all lanes redundantly; lane 0 commits)
    float mx = acc[0]; int gi = 0;
#pragma unroll
    for (int e = 1; e < NEXP; ++e) if (acc[e] > mx) { mx = acc[e]; gi = e; }
    float s = 0.0f;
#pragma unroll
    for (int e = 0; e < NEXP; ++e) s += __expf(acc[e] - mx);
    float pg = 1.0f / s;                   // exp(0)/sum

    if (lane == 0) {
        coeff[t] = SCALING * pg;
        int pos = atomicAdd(&counts[gi], 1);
        order[gi * TOK + pos] = t;
    }
}

// ---------------------------------------------------------------- grouped LoRA GEMM
// one wave32 per 32-token tile (2 M-tiles) of one expert; 4 waves / block
__global__ __launch_bounds__(128)
void moe_lora_gemm(const uint16_t* __restrict__ Xb,
                   const uint16_t* __restrict__ Abf_l,   // [E,R,H] bf16
                   const uint16_t* __restrict__ Bbf_l,   // [E,H,R] bf16
                   const float* __restrict__ coeff,
                   const int* __restrict__ order,
                   const int* __restrict__ counts,
                   float* __restrict__ out)
{
    __shared__ float hbuf[4][32][RANK + 1];   // +1 pad: kill LDS bank conflicts
    __shared__ int   tks[4][32];

    const int lane   = threadIdx.x & 31;
    const int w      = threadIdx.x >> 5;
    const int e      = blockIdx.x >> 6;       // 64 blocks per expert
    const int bgroup = blockIdx.x & 63;
    const int cnt    = counts[e];
    const int wstart = bgroup * 128 + w * 32; // this wave's first token slot
    if (wstart >= cnt) return;                // uniform per wave -> EXEC stays full

    const int m16 = lane & 15;
    const int hi  = lane >> 4;

    {
        int idx = wstart + lane;
        if (idx >= cnt) idx = cnt - 1;        // clamp; stores masked below
        tks[w][lane] = order[e * TOK + idx];
    }
    asm volatile("s_wait_dscnt 0" ::: "memory");

    const uint16_t* Ae = Abf_l + (size_t)e * RANK * HDIM;
    const uint16_t* Be = Bbf_l + (size_t)e * HDIM * RANK;

    const uint16_t* xrow0 = Xb + (size_t)tks[w][m16]      * HDIM;
    const uint16_t* xrow1 = Xb + (size_t)tks[w][16 + m16] * HDIM;

    // ---- phase 1: h[32,64] = X[32,2048] @ A_e^T ----
    v8f cc[2][4] = {};
    for (int k0 = 0; k0 < HDIM; k0 += 32) {
        Frag a0, a1;
        a0.q[0] = *(const uint4*)(xrow0 + k0 + hi * 8);        // K = hi*8 .. +7
        a0.q[1] = *(const uint4*)(xrow0 + k0 + 16 + hi * 8);   // K = 16+hi*8 .. +7
        a1.q[0] = *(const uint4*)(xrow1 + k0 + hi * 8);
        a1.q[1] = *(const uint4*)(xrow1 + k0 + 16 + hi * 8);
#pragma unroll
        for (int rt = 0; rt < 4; ++rt) {
            const uint16_t* bp = Ae + (size_t)(rt * 16 + m16) * HDIM + k0 + hi * 8;
            Frag b;
            b.q[0] = *(const uint4*)(bp);
            b.q[1] = *(const uint4*)(bp + 16);
            cc[0][rt] = wmma_bf16(a0, b, cc[0][rt]);   // B fragment reused by both M-tiles
            cc[1][rt] = wmma_bf16(a1, b, cc[1][rt]);
        }
    }

    // C/D layout (VGPR j -> M = j + hi*8, N = m16) -> LDS
#pragma unroll
    for (int mt = 0; mt < 2; ++mt)
#pragma unroll
        for (int rt = 0; rt < 4; ++rt)
#pragma unroll
            for (int j = 0; j < 8; ++j)
                hbuf[w][mt * 16 + j + hi * 8][rt * 16 + m16] = cc[mt][rt][j];
    asm volatile("s_wait_dscnt 0" ::: "memory");

    // per-row (token) coefficients
    float cf[2][8]; int trow[2][8]; bool vrow[2][8];
#pragma unroll
    for (int mt = 0; mt < 2; ++mt)
#pragma unroll
        for (int j = 0; j < 8; ++j) {
            int m = mt * 16 + j + hi * 8;
            trow[mt][j] = tks[w][m];
            vrow[mt][j] = (wstart + m) < cnt;
            cf[mt][j]   = coeff[trow[mt][j]];
        }

    // re-gather h as A-operand fragments (per M-tile, K = 0..31 and 32..63), bf16
    Frag ha[2][2];
#pragma unroll
    for (int mt = 0; mt < 2; ++mt)
#pragma unroll
        for (int ks = 0; ks < 2; ++ks)
#pragma unroll
            for (int i = 0; i < 16; ++i) {
                int r = ks * 32 + hi * 16 + i;
                ha[mt][ks].v[i] = (__bf16)hbuf[w][mt * 16 + m16][r];
            }

    // ---- phase 2: y[32,2048] = h[32,64] @ B_e^T, scaled, scattered ----
    for (int n0 = 0; n0 < HDIM; n0 += 16) {
        const int col = n0 + m16;
        const uint16_t* bp = Be + (size_t)col * RANK + hi * 16;
        Frag b0, b1;
        b0.q[0] = *(const uint4*)(bp);
        b0.q[1] = *(const uint4*)(bp + 8);
        b1.q[0] = *(const uint4*)(bp + 32);
        b1.q[1] = *(const uint4*)(bp + 40);
        v8f y0 = {}, y1 = {};
        y0 = wmma_bf16(ha[0][0], b0, y0);   // B fragments shared by both M-tiles
        y0 = wmma_bf16(ha[0][1], b1, y0);
        y1 = wmma_bf16(ha[1][0], b0, y1);
        y1 = wmma_bf16(ha[1][1], b1, y1);
#pragma unroll
        for (int j = 0; j < 8; ++j) {
            if (vrow[0][j]) out[(size_t)trow[0][j] * HDIM + col] = y0[j] * cf[0][j];
            if (vrow[1][j]) out[(size_t)trow[1][j] * HDIM + col] = y1[j] * cf[1][j];
        }
    }
}

// ----------------------------------------------------------------
extern "C" void kernel_launch(void* const* d_in, const int* in_sizes, int n_in,
                              void* d_out, int out_size, void* d_ws, size_t ws_size,
                              hipStream_t stream) {
    const float* hs    = (const float*)d_in[0];   // [L+1,B,S,H]
    const float* gates = (const float*)d_in[1];   // [L]
    const float* A     = (const float*)d_in[2];   // [L,E,R,H]
    const float* Bm    = (const float*)d_in[3];   // [L,E,H,R]
    const float* gw    = (const float*)d_in[4];   // [L,E,H]
    float* out = (float*)d_out;                   // [B,S,H] = running x buffer

    const size_t WN = (size_t)LAYERS * NEXP * RANK * HDIM;   // 8,388,608
    char* w = (char*)d_ws;
    uint16_t* Abf  = (uint16_t*)w;  w += WN * 2;
    uint16_t* Bbf  = (uint16_t*)w;  w += WN * 2;
    uint16_t* Xb   = (uint16_t*)w;  w += (size_t)TOK * HDIM * 2;
    float*    cof  = (float*)w;     w += (size_t)TOK * 4;
    int*      ord  = (int*)w;       w += (size_t)NEXP * TOK * 4;
    int*      cnts = (int*)w;       w += 256;

    cvt_bf16<<<2048, 256, 0, stream>>>(A,  Abf, (int)WN);
    cvt_bf16<<<2048, 256, 0, stream>>>(Bm, Bbf, (int)WN);

    for (int i = 0; i < LAYERS; ++i) {
        hipMemsetAsync(cnts, 0, NEXP * sizeof(int), stream);
        const float* xsrc = (i == 0) ? hs : out;
        const float* hx   = hs + (size_t)((i == 0) ? 1 : i) * TOK * HDIM;
        blend_gate_route<<<TOK / 8, 256, 0, stream>>>(
            xsrc, hx, gates, i, gw + (size_t)i * NEXP * HDIM,
            Xb, cof, ord, cnts);
        moe_lora_gemm<<<NEXP * 64, 128, 0, stream>>>(
            Xb, Abf + (size_t)i * NEXP * RANK * HDIM,
            Bbf + (size_t)i * NEXP * HDIM * RANK,
            cof, ord, cnts, out);
    }
}
